// MAG_68083821576842
// MI455X (gfx1250) — compile-verified, matrix operand order checked by reading, not run on hardware
//
#include <hip/hip_runtime.h>

// ---- problem constants (match reference) ----
#define TB   512            // graphs
#define NN   96             // nodes per graph (exact, no padding)
#define NH   8              // heads
#define DHd  32             // head dim  (== WMMA K for f16!)
#define HIDd 256
#define MLPd 256
#define TNODES (TB*NN)      // 49152
#define EDGES 196608
#define NEGB  -60000.0f     // finite stand-in for -1e9 that survives f16

typedef __attribute__((ext_vector_type(16))) _Float16 v16h;
typedef __attribute__((ext_vector_type(8)))  float    v8f;

// ---------------- elementwise prep kernels ----------------

__global__ void k_convert_f16(const float* __restrict__ s, _Float16* __restrict__ d, int n) {
  int i = blockIdx.x * blockDim.x + threadIdx.x;
  if (i < n) d[i] = (_Float16)s[i];
}

// s[rows][cols] f32  ->  d[cols][rows] f16   (weight transpose: [in][out] -> [out][in])
__global__ void k_transpose_f16(const float* __restrict__ s, _Float16* __restrict__ d,
                                int rows, int cols) {
  int i = blockIdx.x * blockDim.x + threadIdx.x;
  if (i < rows * cols) {
    int r = i / cols, c = i % cols;
    d[(size_t)c * rows + r] = (_Float16)s[i];
  }
}

// combined bias init: NEG everywhere, 0 on the diagonal (self-loops)
__global__ void k_bias_fill(_Float16* __restrict__ bias) {
  int i = blockIdx.x * blockDim.x + threadIdx.x;
  const int total = TB * NH * NN * NN;
  if (i < total) {
    int qk = i % (NN * NN);
    bias[i] = (qk / NN == qk % NN) ? (_Float16)0.0f : (_Float16)NEGB;
  }
}

// per-edge: bias_flat = edge_attr @ W_bond + b_bond, scattered dense
__global__ void k_edge_scatter(const float* __restrict__ ea, const float* __restrict__ Wb,
                               const float* __restrict__ bb, const int* __restrict__ eidx,
                               _Float16* __restrict__ bias) {
  int e = blockIdx.x * blockDim.x + threadIdx.x;
  if (e >= EDGES) return;
  int s  = eidx[e];
  int dn = eidx[EDGES + e];
  int b  = s / NN, ls = s % NN, ld = dn % NN;
  float a[7];
#pragma unroll
  for (int d = 0; d < 7; d++) a[d] = ea[(size_t)e * 7 + d];
#pragma unroll
  for (int h = 0; h < NH; h++) {
    float v = bb[h];
#pragma unroll
    for (int d = 0; d < 7; d++) v += a[d] * Wb[d * NH + h];
    bias[(((size_t)b * NH + h) * NN + ls) * NN + ld] = (_Float16)v;
  }
}

// ---------------- register-blocked WMMA GEMM ----------------
//   C = act(A[M,K] * Bt[N,K]^T + bias)
// one wave computes a 16x64 strip: A fragment loaded once per k-step,
// 4 WMMAs per k-step against 4 B tiles (load:wmma = 5:4, A traffic /4).
// block = 4 waves = 4 consecutive m-tiles sharing the same 64-col strip.
// MODE 0: C f16 row-major [M][Ncols]
// MODE 1: q/k head layout  [b][h][n][d]   (row=b*96+n, col=h*32+d)
// MODE 2: v transposed     [b][h][d][n]
template <int MODE, int RELU>
__global__ __launch_bounds__(128) void k_gemm(const _Float16* __restrict__ A,
                                              const _Float16* __restrict__ Bt,
                                              const float* __restrict__ bias,
                                              _Float16* __restrict__ C,
                                              int M, int Ncols, int K) {
  const int lane = threadIdx.x & 31;
  const int wave = threadIdx.x >> 5;
  const int tm = blockIdx.x * 4 + wave;   // 16-row tile per wave
  const int n0 = blockIdx.y * 64;         // 64-col strip per block
  const int half = lane >> 4;             // 0: k 0..15, 1: k 16..31
  const int l16  = lane & 15;
  const size_t aoff = (size_t)(tm * 16 + l16) * K + half * 16;
  const size_t boff = (size_t)(n0 + l16) * K + half * 16;
  const size_t bstr = (size_t)16 * K;     // 16 cols of Bt

  v8f acc[4] = {};
  for (int kk = 0; kk < K; kk += 32) {
    v16h a  = *(const v16h*)(A + aoff + kk);
    __builtin_prefetch(A + aoff + kk + 256, 0, 1);   // global_prefetch_b8
    v16h b0 = *(const v16h*)(Bt + boff + 0 * bstr + kk);
    v16h b1 = *(const v16h*)(Bt + boff + 1 * bstr + kk);
    v16h b2 = *(const v16h*)(Bt + boff + 2 * bstr + kk);
    v16h b3 = *(const v16h*)(Bt + boff + 3 * bstr + kk);
    acc[0] = __builtin_amdgcn_wmma_f32_16x16x32_f16(false, a, false, b0, (short)0, acc[0], false, false);
    acc[1] = __builtin_amdgcn_wmma_f32_16x16x32_f16(false, a, false, b1, (short)0, acc[1], false, false);
    acc[2] = __builtin_amdgcn_wmma_f32_16x16x32_f16(false, a, false, b2, (short)0, acc[2], false, false);
    acc[3] = __builtin_amdgcn_wmma_f32_16x16x32_f16(false, a, false, b3, (short)0, acc[3], false, false);
  }

#pragma unroll
  for (int j = 0; j < 4; j++) {
    const int col = n0 + j * 16 + l16;
    const float bv = bias ? bias[col] : 0.0f;
#pragma unroll
    for (int r = 0; r < 8; r++) {
      int row = tm * 16 + r + half * 8;    // C layout: vgpr r, lanes>=16 hold M+8
      float v = acc[j][r] + bv;
      if (RELU) v = v > 0.0f ? v : 0.0f;
      if (MODE == 0) {
        C[(size_t)row * Ncols + col] = (_Float16)v;
      } else {
        int b = row / NN, n = row % NN;
        int hh = col >> 5, d = col & 31;
        if (MODE == 1) C[(((size_t)b * NH + hh) * NN + n) * DHd + d] = (_Float16)v;
        else           C[(((size_t)b * NH + hh) * DHd + d) * NN + n] = (_Float16)v;
      }
    }
  }
}

// ---------------- fused masked attention: scores -> softmax -> attn@V ----------------
// one wave per (graph b, head h, 16-row q tile); 6 score WMMAs + 6 ctx WMMAs
__global__ __launch_bounds__(128) void k_attn(const _Float16* __restrict__ q16,
                                              const _Float16* __restrict__ k16,
                                              const _Float16* __restrict__ v16t,
                                              const _Float16* __restrict__ bias16,
                                              _Float16* __restrict__ ctx16) {
  __shared__ __align__(32) _Float16 lds[4][16][NN];   // per-wave attn tile (A layout bounce)
  const int lane = threadIdx.x & 31;
  const int wave = threadIdx.x >> 5;
  const int wid  = blockIdx.x * 4 + wave;
  const int qt = wid % (NN / 16);
  const int h  = (wid / (NN / 16)) % NH;
  const int b  = wid / ((NN / 16) * NH);
  const int half = lane >> 4, l16 = lane & 15;

  const _Float16* Qb = q16 + ((size_t)b * NH + h) * NN * DHd;
  const _Float16* Kb = k16 + ((size_t)b * NH + h) * NN * DHd;
  const _Float16* Vb = v16t + ((size_t)b * NH + h) * DHd * NN;
  const _Float16* Bb = bias16 + ((size_t)b * NH + h) * NN * NN;

  // A operand: Q rows qt*16.. (reused across all 6 k-tiles; DH=32 == WMMA K)
  v16h aQ = *(const v16h*)(Qb + (size_t)(qt * 16 + l16) * DHd + half * 16);

  const float scale = 0.17677669529663687f;           // 1/sqrt(32)
  float sv[6][8];
#pragma unroll
  for (int kt = 0; kt < 6; kt++) {
    v16h bK = *(const v16h*)(Kb + (size_t)(kt * 16 + l16) * DHd + half * 16);
    v8f s = {};
    s = __builtin_amdgcn_wmma_f32_16x16x32_f16(false, aQ, false, bK, (short)0, s,
                                               false, false);
#pragma unroll
    for (int r = 0; r < 8; r++) {
      int qr = qt * 16 + r + half * 8;
      int kc = kt * 16 + l16;
      sv[kt][r] = s[r] * scale + (float)Bb[(size_t)qr * NN + kc];
    }
  }

  // softmax over each row: row = (r, half); elements spread over 16 lanes x 6 tiles.
  // xor-shuffles with masks 1,2,4,8 stay inside each 16-lane half.
#pragma unroll
  for (int r = 0; r < 8; r++) {
    float m = -1e30f;
#pragma unroll
    for (int kt = 0; kt < 6; kt++) m = fmaxf(m, sv[kt][r]);
    for (int off = 1; off < 16; off <<= 1) m = fmaxf(m, __shfl_xor(m, off, 32));
    float sum = 0.0f;
#pragma unroll
    for (int kt = 0; kt < 6; kt++) { float e = __expf(sv[kt][r] - m); sv[kt][r] = e; sum += e; }
    for (int off = 1; off < 16; off <<= 1) sum += __shfl_xor(sum, off, 32);
    float inv = 1.0f / sum;
    int rr = r + half * 8;
#pragma unroll
    for (int kt = 0; kt < 6; kt++)
      lds[wave][rr][kt * 16 + l16] = (_Float16)(sv[kt][r] * inv);
  }
  __syncthreads();

  // ctx tile = attn[16 x 96] @ V[96 x 32]; A from LDS, B from transposed V (contig k)
#pragma unroll
  for (int nt = 0; nt < 2; nt++) {
    v8f c = {};
#pragma unroll
    for (int kc = 0; kc < 3; kc++) {
      v16h a = *(const v16h*)(&lds[wave][l16][kc * 32 + half * 16]);
      v16h bv = *(const v16h*)(Vb + (size_t)(nt * 16 + l16) * NN + kc * 32 + half * 16);
      c = __builtin_amdgcn_wmma_f32_16x16x32_f16(false, a, false, bv, (short)0, c,
                                                 false, false);
    }
#pragma unroll
    for (int r = 0; r < 8; r++) {
      int qr = qt * 16 + r + half * 8;
      int t = b * NN + qr;
      int col = h * DHd + nt * 16 + l16;
      ctx16[(size_t)t * HIDd + col] = (_Float16)c[r];
    }
  }
}

// masked mean pool over nodes (mean commutes with Wo, so pool before Wo)
__global__ void k_pool(const _Float16* __restrict__ ctx16, _Float16* __restrict__ pool16) {
  int b = blockIdx.x, c = threadIdx.x;   // 256 threads
  float s = 0.0f;
  for (int n = 0; n < NN; n++) s += (float)ctx16[((size_t)b * NN + n) * HIDd + c];
  pool16[(size_t)b * HIDd + c] = (_Float16)(s * (1.0f / NN));
}

// final tiny projection: logits = y2 @ W_out2 + b_out2
__global__ void k_head(const _Float16* __restrict__ y2, const float* __restrict__ W2,
                       const float* __restrict__ b2, float* __restrict__ out) {
  int g = blockIdx.x * blockDim.x + threadIdx.x;
  if (g < TB) {
    float s = b2[0];
    for (int k = 0; k < HIDd; k++) s += (float)y2[(size_t)g * HIDd + k] * W2[k];
    out[g] = s;
  }
}

// ---------------- launch ----------------

extern "C" void kernel_launch(void* const* d_in, const int* in_sizes, int n_in,
                              void* d_out, int out_size, void* d_ws, size_t ws_size,
                              hipStream_t stream) {
  const float* node_features = (const float*)d_in[0];
  const float* edge_attr     = (const float*)d_in[1];
  const float* W_in1 = (const float*)d_in[2];  const float* b_in1 = (const float*)d_in[3];
  const float* W_in2 = (const float*)d_in[4];  const float* b_in2 = (const float*)d_in[5];
  const float* W_bond = (const float*)d_in[6]; const float* b_bond = (const float*)d_in[7];
  const float* Wq = (const float*)d_in[8];
  const float* Wk = (const float*)d_in[9];
  const float* Wv = (const float*)d_in[10];
  const float* Wo = (const float*)d_in[11];
  const float* W_out1 = (const float*)d_in[12]; const float* b_out1 = (const float*)d_in[13];
  const float* W_out2 = (const float*)d_in[14]; const float* b_out2 = (const float*)d_in[15];
  const int* edge_index = (const int*)d_in[16];
  float* out = (float*)d_out;

  // workspace carve-out (all f16)
  char* p = (char*)d_ws;
  auto carve = [&](size_t elems) -> _Float16* {
    _Float16* r = (_Float16*)p;
    size_t bytes = (elems * sizeof(_Float16) + 255) & ~(size_t)255;
    p += bytes;
    return r;
  };
  _Float16* x16    = carve((size_t)TNODES * 32);
  _Float16* Wi1T   = carve(256 * 32);
  _Float16* Wi2T   = carve(256 * 256);
  _Float16* WqT    = carve(256 * 256);
  _Float16* WkT    = carve(256 * 256);
  _Float16* WvT    = carve(256 * 256);
  _Float16* WoT    = carve(256 * 256);
  _Float16* Wo1T   = carve(256 * 256);
  _Float16* hid16  = carve((size_t)TNODES * MLPd);
  _Float16* h16    = carve((size_t)TNODES * HIDd);
  _Float16* q16    = carve((size_t)TNODES * HIDd);
  _Float16* k16    = carve((size_t)TNODES * HIDd);
  _Float16* v16t   = carve((size_t)TNODES * HIDd);
  _Float16* bias16 = carve((size_t)TB * NH * NN * NN);
  _Float16* ctx16  = carve((size_t)TNODES * HIDd);
  _Float16* pool16 = carve((size_t)TB * HIDd);
  _Float16* y1     = carve((size_t)TB * HIDd);
  _Float16* y2     = carve((size_t)TB * HIDd);

  const int THR = 256;
  // prep: conversions + weight transposes
  k_convert_f16<<<(TNODES * 32 + THR - 1) / THR, THR, 0, stream>>>(node_features, x16, TNODES * 32);
  k_transpose_f16<<<(32 * 256 + THR - 1) / THR, THR, 0, stream>>>(W_in1, Wi1T, 32, 256);
  k_transpose_f16<<<(256 * 256 + THR - 1) / THR, THR, 0, stream>>>(W_in2, Wi2T, 256, 256);
  k_transpose_f16<<<(256 * 256 + THR - 1) / THR, THR, 0, stream>>>(Wq, WqT, 256, 256);
  k_transpose_f16<<<(256 * 256 + THR - 1) / THR, THR, 0, stream>>>(Wk, WkT, 256, 256);
  k_transpose_f16<<<(256 * 256 + THR - 1) / THR, THR, 0, stream>>>(Wv, WvT, 256, 256);
  k_transpose_f16<<<(256 * 256 + THR - 1) / THR, THR, 0, stream>>>(Wo, WoT, 256, 256);
  k_transpose_f16<<<(256 * 256 + THR - 1) / THR, THR, 0, stream>>>(W_out1, Wo1T, 256, 256);

  // dense bond bias
  {
    int total = TB * NH * NN * NN;
    k_bias_fill<<<(total + THR - 1) / THR, THR, 0, stream>>>(bias16);
    k_edge_scatter<<<(EDGES + THR - 1) / THR, THR, 0, stream>>>(edge_attr, W_bond, b_bond,
                                                                edge_index, bias16);
  }

  // node MLP + QKV projections (register-blocked WMMA GEMMs; M,N multiples of 64)
  dim3 blk(128);
  k_gemm<0, 1><<<dim3(TNODES / 64, MLPd / 64), blk, 0, stream>>>(x16, Wi1T, b_in1, hid16,
                                                                 TNODES, MLPd, 32);
  k_gemm<0, 0><<<dim3(TNODES / 64, HIDd / 64), blk, 0, stream>>>(hid16, Wi2T, b_in2, h16,
                                                                 TNODES, HIDd, MLPd);
  k_gemm<1, 0><<<dim3(TNODES / 64, HIDd / 64), blk, 0, stream>>>(h16, WqT, nullptr, q16,
                                                                 TNODES, HIDd, HIDd);
  k_gemm<1, 0><<<dim3(TNODES / 64, HIDd / 64), blk, 0, stream>>>(h16, WkT, nullptr, k16,
                                                                 TNODES, HIDd, HIDd);
  k_gemm<2, 0><<<dim3(TNODES / 64, HIDd / 64), blk, 0, stream>>>(h16, WvT, nullptr, v16t,
                                                                 TNODES, HIDd, HIDd);

  // fused attention (scores + bias -> softmax -> attn@V)
  k_attn<<<dim3(TB * NH * (NN / 16) / 4), blk, 0, stream>>>(q16, k16, v16t, bias16, ctx16);

  // pool, then Wo on pooled (mean/Wo commute), output MLP
  k_pool<<<dim3(TB), dim3(HIDd), 0, stream>>>(ctx16, pool16);
  k_gemm<0, 0><<<dim3(TB / 64, HIDd / 64), blk, 0, stream>>>(pool16, WoT, nullptr, y1,
                                                             TB, HIDd, HIDd);
  k_gemm<0, 1><<<dim3(TB / 64, HIDd / 64), blk, 0, stream>>>(y1, Wo1T, b_out1, y2,
                                                             TB, HIDd, HIDd);
  k_head<<<dim3(2), dim3(256), 0, stream>>>(y2, W_out2, b_out2, out);
}
